// NeuralSplineCoupling_61787399520624
// MI455X (gfx1250) — compile-verified, matrix operand order checked by reading, not run on hardware
//
#include <hip/hip_runtime.h>

// ---------------- problem constants (from reference) ----------------
#define KNOTS   16
#define SD      47          // 3*KNOTS-1
#define DFULL   64
#define DHALF   32
#define CCOND   16
#define CIN     48
#define CINP    64          // padded K for GEMM0
#define H1      128
#define H2      128
#define NOUT    1504        // DHALF * SD
#define NTILE3  94          // NOUT / 16
#define MT      16          // rows per workgroup
#define NTHREADS 128
#define NWAVES  4
#define BN_EPS  1e-5f

typedef __attribute__((ext_vector_type(16))) _Float16 v16h;
typedef __attribute__((ext_vector_type(8)))  float    v8f;

// ---------------- workspace (d_ws) layout: pre-converted f16 weights --------
// w0t: f16 [128][64]   bytes [0, 16384)          (K padded 48->64 w/ zeros)
// w1t: f16 [128][128]  bytes [16384, 49152)
// w2t: f16 [1504][128] bytes [49152, 434176)     (tile t = 4KB contiguous)
#define WS_W0T_B  0
#define WS_W1T_B  16384
#define WS_W2T_B  49152
#define E0 (H1 * CINP)      // 8192
#define E1 (H2 * H1)      // 16384
#define E2 (NOUT * H2)      // 192512

// ---------------- LDS layout (bytes, 128-aligned offsets) -------------------
// [SM_W1T, +32768) holds W1^T during GEMM0/1, then is reused as 4 waves x
// 2 double-buffered W2 chunks (f16 [16][128] = 4096 B each) during GEMM2.
#define SM_W0T   0          // f16 [128][64]  = 16384
#define SM_B0    16384      // 128 f32
#define SM_B1    16896      // 128 f32
#define SM_W1T   17408      // f16 [128][128] = 32768 ; -> W2 chunk buffers
#define SM_H0    50176      // f16 [16][64]   = 2048
#define SM_H1    52224      // f16 [16][128]  = 4096
#define SM_H2    56320      // f16 [16][128]  = 4096
#define SM_B2    60416      // 1504 f32 = 6016
#define SM_P     66432      // f32 [16][1504] = 96256
#define SM_LD    162688     // 16 f32
#define SM_TOTAL 162752

#define WMMA_F16(a, b, c) \
  __builtin_amdgcn_wmma_f32_16x16x32_f16(false, (a), false, (b), (short)0, (c), false, false)

#if __has_builtin(__builtin_amdgcn_global_load_async_to_lds_b128) && \
    __has_builtin(__builtin_amdgcn_s_wait_asynccnt)
#define USE_ASYNC 1
#else
#define USE_ASYNC 0
#endif

// pointer types the async builtin expects: v4i in explicit address spaces
typedef __attribute__((__vector_size__(16))) int v4i_t;
typedef __attribute__((address_space(1))) v4i_t glb_v4i;
typedef __attribute__((address_space(3))) v4i_t lds_v4i;

// async DMA: 16 bytes global -> LDS, tracked by ASYNCcnt (CDNA5 path)
__device__ __forceinline__ void async_b128(const void* g, void* l) {
#if USE_ASYNC
  __builtin_amdgcn_global_load_async_to_lds_b128((glb_v4i*)g, (lds_v4i*)l, 0, 0);
#else
  *(uint4*)l = *(const uint4*)g;
#endif
}
__device__ __forceinline__ void wait_async_0() {
#if USE_ASYNC
  __builtin_amdgcn_s_wait_asynccnt(0);
#endif
  __asm__ volatile("" ::: "memory");
}
__device__ __forceinline__ void wait_async_8() {
#if USE_ASYNC
  __builtin_amdgcn_s_wait_asynccnt(8);
#endif
  __asm__ volatile("" ::: "memory");
}

// CDNA5 16-bit A/B fragment: lanes 0-15 hold K {0..7,16..23}, lanes 16-31 hold
// K {8..15,24..31} (+32*ktile); dwords pack consecutive-K f16 pairs.
__device__ __forceinline__ v16h load_frag(const _Float16* rowptr, int ktile, int lane) {
  const int kh = (lane >> 4) & 1;
  const unsigned int* b = (const unsigned int*)(rowptr + ktile * 32 + kh * 8);
  union { v16h v; unsigned int u[8]; } r;
  r.u[0] = b[0];  r.u[1] = b[1];  r.u[2] = b[2];  r.u[3] = b[3];
  r.u[4] = b[8];  r.u[5] = b[9];  r.u[6] = b[10]; r.u[7] = b[11];
  return r.v;
}

__device__ __forceinline__ float swishf(float x) {
  return __fdividef(x, 1.0f + __expf(-x));
}

// ---------------- prologue: one-shot f32 -> f16 transposed weight convert ---
__global__ __launch_bounds__(256)
void convert_weights(const float* __restrict__ W0, const float* __restrict__ W1,
                     const float* __restrict__ W2, _Float16* __restrict__ ws) {
  int idx = blockIdx.x * 256 + threadIdx.x;
  if (idx < E0) {                               // w0t[n][k], k padded
    int n = idx >> 6, k = idx & 63;
    ws[idx] = (_Float16)((k < CIN) ? W0[k * H1 + n] : 0.0f);
  } else if (idx < E0 + E1) {                   // w1t[n][k]
    int j = idx - E0; int n = j >> 7, k = j & 127;
    ws[idx] = (_Float16)W1[k * H2 + n];
  } else if (idx < E0 + E1 + E2) {              // w2t[n][k]
    int j = idx - (E0 + E1); int n = j >> 7, k = j & 127;
    ws[idx] = (_Float16)W2[k * NOUT + n];
  }
}

__global__ __launch_bounds__(NTHREADS)
void nsc_fused_kernel(const float* __restrict__ xg,   const float* __restrict__ cg,
                      const float* __restrict__ bnsc, const float* __restrict__ bnbi,
                      const float* __restrict__ bnmu, const float* __restrict__ bnvr,
                      const float* __restrict__ b0,   const float* __restrict__ b1,
                      const float* __restrict__ b2,   const _Float16* __restrict__ ws,
                      float* __restrict__ out, int Nrows) {
  __shared__ __align__(16) unsigned char smem[SM_TOTAL];
  _Float16* sh0 = (_Float16*)(smem + SM_H0);
  _Float16* sh1 = (_Float16*)(smem + SM_H1);
  _Float16* sh2 = (_Float16*)(smem + SM_H2);
  float* sb0 = (float*)(smem + SM_B0);
  float* sb1 = (float*)(smem + SM_B1);
  float* sb2 = (float*)(smem + SM_B2);
  float* sp  = (float*)(smem + SM_P);
  float* sld = (float*)(smem + SM_LD);

  const int tid  = threadIdx.x;
  const int lane = tid & 31;
  const int wave = tid >> 5;
  const int row0 = blockIdx.x * MT;
  const char* wsb = (const char*)ws;

  // -------- phase 0: async-stage f16 weights; BN'd input tile; biases -------
  for (int off = tid * 16; off < 16384; off += NTHREADS * 16)          // w0t
    async_b128(wsb + WS_W0T_B + off, smem + SM_W0T + off);
  for (int off = tid * 16; off < 32768; off += NTHREADS * 16)          // w1t
    async_b128(wsb + WS_W1T_B + off, smem + SM_W1T + off);

  if (tid < H1) { sb0[tid] = b0[tid]; sb1[tid] = b1[tid]; }
  if (tid < MT) sld[tid] = 0.0f;
  for (int i = tid; i < NOUT; i += NTHREADS) sb2[i] = b2[i];

  for (int idx = tid; idx < MT * CINP; idx += NTHREADS) {
    int r = idx >> 6, j = idx & 63;
    float v = 0.0f;
    if (j < CIN) {
      int grow = row0 + r;
      float f = (j < DHALF) ? xg[grow * DFULL + DHALF + j]
                            : cg[grow * CCOND + (j - DHALF)];
      v = (f - bnmu[j]) * __frsqrt_rn(bnvr[j] + BN_EPS) * bnsc[j] + bnbi[j];
    }
    sh0[idx] = (_Float16)v;
  }
  wait_async_0();
  __syncthreads();

  const int frow  = lane & 15;
  const int mbase = (lane >> 4) * 8;
  const _Float16* sw0t = (const _Float16*)(smem + SM_W0T);
  const _Float16* sw1t = (const _Float16*)(smem + SM_W1T);

  // -------- GEMM0: h1 = swish(h0 @ W0 + b0) ---------------------------------
  #pragma unroll
  for (int cti = 0; cti < 2; ++cti) {
    const int ct = wave * 2 + cti;
    v8f acc = {};
    #pragma unroll
    for (int kt = 0; kt < 2; ++kt) {
      v16h a = load_frag(sh0  + frow * CINP, kt, lane);
      v16h b = load_frag(sw0t + (ct * 16 + frow) * CINP, kt, lane);
      acc = WMMA_F16(a, b, acc);
    }
    const int n = ct * 16 + frow;
    const float bias = sb0[n];
    #pragma unroll
    for (int v = 0; v < 8; ++v)
      sh1[(mbase + v) * H1 + n] = (_Float16)swishf(acc[v] + bias);
  }
  __syncthreads();

  // -------- GEMM1: h2 = swish(h1 @ W1 + b1) ---------------------------------
  #pragma unroll
  for (int cti = 0; cti < 2; ++cti) {
    const int ct = wave * 2 + cti;
    v8f acc = {};
    #pragma unroll
    for (int kt = 0; kt < 4; ++kt) {
      v16h a = load_frag(sh1  + frow * H1, kt, lane);
      v16h b = load_frag(sw1t + (ct * 16 + frow) * H1, kt, lane);
      acc = WMMA_F16(a, b, acc);
    }
    const int n = ct * 16 + frow;
    const float bias = sb1[n];
    #pragma unroll
    for (int v = 0; v < 8; ++v)
      sh2[(mbase + v) * H2 + n] = (_Float16)swishf(acc[v] + bias);
  }
  __syncthreads();   // W1t region now dead -> reuse as per-wave W2 chunks

  // -------- GEMM2: p = h2 @ W2 + b2 (async double-buffered W2 tiles) --------
  const char* w2g = wsb + WS_W2T_B;                 // tile t at +t*4096, 4KB
  char* cbuf0 = (char*)smem + SM_W1T + wave * 8192;
  char* cbuf1 = cbuf0 + 4096;
  {
    int t0 = wave;
    #pragma unroll
    for (int i = 0; i < 8; ++i)                     // preload first tile
      async_b128(w2g + t0 * 4096 + i * 512 + lane * 16,
                 cbuf0 + i * 512 + lane * 16);
  }
  int cur = 0;
  for (int t = wave; t < NTILE3; t += NWAVES) {
    const int tn = t + NWAVES;
    char* nbuf = cur ? cbuf0 : cbuf1;
    if (tn < NTILE3) {
      #pragma unroll
      for (int i = 0; i < 8; ++i)                   // DMA next tile
        async_b128(w2g + tn * 4096 + i * 512 + lane * 16,
                   nbuf + i * 512 + lane * 16);
      wait_async_8();                               // current tile landed
    } else {
      wait_async_0();
    }
    const _Float16* chunk = (const _Float16*)(cur ? cbuf1 : cbuf0);
    v8f acc = {};
    #pragma unroll
    for (int kt = 0; kt < 4; ++kt) {
      v16h a = load_frag(sh2   + frow * H2, kt, lane);
      v16h b = load_frag(chunk + frow * H2, kt, lane);
      acc = WMMA_F16(a, b, acc);
    }
    const int n = t * 16 + frow;
    const float bias = sb2[n];
    #pragma unroll
    for (int v = 0; v < 8; ++v)
      sp[(mbase + v) * NOUT + n] = acc[v] + bias;
    cur ^= 1;
  }
  __syncthreads();

  // -------- spline phase: 16 rows x 32 dims ---------------------------------
  for (int pid = tid; pid < MT * DHALF; pid += NTHREADS) {
    const int row = pid & 15;
    const int dim = pid >> 4;
    const int grow = row0 + row;
    const float* pr = sp + row * NOUT + dim * SD;

    float wv[KNOTS], hv[KNOTS], sv[KNOTS - 1];
    float mx = -3.0e38f;
    #pragma unroll
    for (int i = 0; i < KNOTS; ++i) { wv[i] = pr[i]; mx = fmaxf(mx, wv[i]); }
    float sum = 0.0f;
    #pragma unroll
    for (int i = 0; i < KNOTS; ++i) { wv[i] = __expf(wv[i] - mx); sum += wv[i]; }
    float inv = __fdividef(1.0f, sum);
    #pragma unroll
    for (int i = 0; i < KNOTS; ++i) wv[i] *= inv;

    mx = -3.0e38f;
    #pragma unroll
    for (int i = 0; i < KNOTS; ++i) { hv[i] = pr[KNOTS + i]; mx = fmaxf(mx, hv[i]); }
    sum = 0.0f;
    #pragma unroll
    for (int i = 0; i < KNOTS; ++i) { hv[i] = __expf(hv[i] - mx); sum += hv[i]; }
    inv = __fdividef(1.0f, sum);
    #pragma unroll
    for (int i = 0; i < KNOTS; ++i) hv[i] *= inv;

    #pragma unroll
    for (int i = 0; i < KNOTS - 1; ++i) {
      float v = pr[2 * KNOTS + i];
      sv[i] = (v > 20.0f) ? v : log1pf(__expf(v));
    }

    const float t  = xg[grow * DFULL + dim];
    const bool inb = (t >= 0.0f) && (t <= 1.0f);
    const float tc = fminf(fmaxf(t, 0.0f), 1.0f);

    int cnt = 0; float cum = 0.0f;
    #pragma unroll
    for (int i = 0; i < KNOTS; ++i) { cum += wv[i]; cnt += (cum <= tc) ? 1 : 0; }
    const int bidx = (cnt > KNOTS - 1) ? (KNOTS - 1) : cnt;

    float xkb = 0.0f, ykb = 0.0f;
    #pragma unroll
    for (int i = 0; i < KNOTS; ++i)
      if (i < bidx) { xkb += wv[i]; ykb += hv[i]; }

    const float dxb = wv[bidx], dyb = hv[bidx];
    const float d0 = (bidx == 0)         ? 1.0f : sv[bidx - 1];
    const float d1 = (bidx == KNOTS - 1) ? 1.0f : sv[bidx];

    const float rdx  = __fdividef(1.0f, dxb);
    const float xi   = (tc - xkb) * rdx;
    const float s    = dyb * rdx;
    const float xi1m = xi * (1.0f - xi);
    const float den  = s + (d0 + d1 - 2.0f * s) * xi1m;
    const float rden = __fdividef(1.0f, den);
    const float yin  = ykb + dyb * (s * xi * xi + d0 * xi1m) * rden;
    const float omxi = 1.0f - xi;
    const float dydx = s * s * (d1 * xi * xi + 2.0f * s * xi1m + d0 * omxi * omxi)
                       * rden * rden;

    out[grow * DFULL + dim] = inb ? yin : t;
    if (inb) atomicAdd(&sld[row], __logf(dydx));   // ds_add_f32
  }

  for (int idx = tid; idx < MT * DHALF; idx += NTHREADS) {
    int r = idx >> 5, j = idx & 31;
    out[(row0 + r) * DFULL + DHALF + j] = xg[(row0 + r) * DFULL + DHALF + j];
  }
  __syncthreads();
  if (tid < MT)
    out[Nrows * DFULL + row0 + tid] = sld[tid];
}

extern "C" void kernel_launch(void* const* d_in, const int* in_sizes, int n_in,
                              void* d_out, int out_size, void* d_ws, size_t ws_size,
                              hipStream_t stream) {
  (void)n_in; (void)out_size; (void)ws_size;
  const float* x    = (const float*)d_in[0];
  const float* c    = (const float*)d_in[1];
  const float* bnsc = (const float*)d_in[2];
  const float* bnbi = (const float*)d_in[3];
  const float* bnmu = (const float*)d_in[4];
  const float* bnvr = (const float*)d_in[5];
  const float* W0   = (const float*)d_in[6];
  const float* b0   = (const float*)d_in[7];
  const float* W1   = (const float*)d_in[8];
  const float* b1   = (const float*)d_in[9];
  const float* W2   = (const float*)d_in[10];
  const float* b2   = (const float*)d_in[11];
  float* out = (float*)d_out;
  _Float16* ws = (_Float16*)d_ws;

  const int Nrows = in_sizes[0] / DFULL;                 // 65536
  const int convBlocks = (E0 + E1 + E2 + 255) / 256;     // 848
  convert_weights<<<convBlocks, 256, 0, stream>>>(W0, W1, W2, ws);
  nsc_fused_kernel<<<Nrows / MT, NTHREADS, 0, stream>>>(
      x, c, bnsc, bnbi, bnmu, bnvr, b0, b1, b2, ws, out, Nrows);
}
